// RSSM_84464826843686
// MI455X (gfx1250) — compile-verified, matrix operand order checked by reading, not run on hardware
//
#include <hip/hip_runtime.h>
#include <math.h>

#define FF   256
#define LAT  128
#define ACTD 8
#define VIS  512
#define BB   512
#define TT   128
#define KGIN 160     // LAT+ACT=136 zero-padded to a multiple of 32
#define GRU3 768

typedef __attribute__((ext_vector_type(16))) __bf16 bf16x16;
typedef __attribute__((ext_vector_type(8)))  __bf16 bf16x8;
typedef __attribute__((ext_vector_type(8)))  float  f32x8;
typedef __attribute__((ext_vector_type(4)))  float  f32x4;

// ---------------------------------------------------------------------------
// Async global->LDS copy (CDNA5: GLOBAL_LOAD_ASYNC_TO_LDS_B128, ASYNCcnt).
// Generic pointers to LDS carry the LDS byte offset in the low 32 bits.
// ---------------------------------------------------------------------------
static __device__ __forceinline__ void async_copy_b128(__bf16* ldst, const __bf16* gsrc) {
  unsigned lds = (unsigned)(uintptr_t)ldst;
  asm volatile("global_load_async_to_lds_b128 %0, %1, off"
               :: "v"(lds), "v"(gsrc) : "memory");
}
static __device__ __forceinline__ void async_wait0() {
  asm volatile("s_wait_asynccnt 0" ::: "memory");
}

// ---------------------------------------------------------------------------
// WMMA fragment helpers (wave32, v_wmma_f32_16x16x32_bf16)
// A (16xK bf16, row-major): lane&15 = row, lane>>4 selects K-half; each lane
// reads two contiguous 8-element (16B) chunks at +0 and +16 elements.
// B (NxK weight, row-major = B^T of the math): same pattern with lane&15 = n.
// C/D (16x16 f32): col = lane&15, rows = (lane>>4)*8 .. +7 over the 8 accs.
// ---------------------------------------------------------------------------
static __device__ __forceinline__ bf16x16 ldfrag(const __bf16* p) {
  bf16x8 c0 = *(const bf16x8*)(p);
  bf16x8 c1 = *(const bf16x8*)(p + 16);
  return __builtin_shufflevector(c0, c1, 0,1,2,3,4,5,6,7,8,9,10,11,12,13,14,15);
}

// OUT[16 x N] (fp32 LDS, stride N) = A[16 x K] (bf16 LDS) * W[N x K]^T + bias
// N-tiles distributed over the 8 waves of the block.
static __device__ __forceinline__ void gemm_tile16(
    const __bf16* __restrict__ sA, int K,
    const __bf16* __restrict__ W, const float* __restrict__ bias,
    float* __restrict__ out, int N, int wave, int lane)
{
  const int l15 = lane & 15;
  const int kh  = (lane >> 4) << 3;
  const int rb  = (lane >> 4) << 3;
  const __bf16* pa = sA + l15 * K + kh;
  for (int n0 = wave * 16; n0 < N; n0 += 8 * 16) {
    const __bf16* pb = W + (size_t)(n0 + l15) * K + kh;
    f32x8 acc = {};
    for (int k0 = 0; k0 < K; k0 += 32) {
      bf16x16 a = ldfrag(pa + k0);
      bf16x16 b = ldfrag(pb + k0);
      acc = __builtin_amdgcn_wmma_f32_16x16x32_bf16(
          false, a, false, b, (short)0, acc, false, false);
    }
    const float bv = bias[n0 + l15];
    #pragma unroll
    for (int v = 0; v < 8; ++v)
      out[(size_t)(rb + v) * N + n0 + l15] = acc[v] + bv;
  }
}

// ---------------------------------------------------------------------------
// Weight fp32 -> bf16 repack
// ---------------------------------------------------------------------------
__global__ void cvt_bf16(const float* __restrict__ s, __bf16* __restrict__ d, int n) {
  int i = blockIdx.x * 256 + threadIdx.x;
  if (i < n) d[i] = (__bf16)s[i];
}

__global__ void cvt_gin(const float* __restrict__ s, __bf16* __restrict__ d) {
  int i = blockIdx.x * 256 + threadIdx.x;
  if (i < FF * KGIN) {
    int r = i / KGIN, k = i % KGIN;
    d[i] = (k < LAT + ACTD) ? (__bf16)s[r * (LAT + ACTD) + k] : (__bf16)0.f;
  }
}

// ---------------------------------------------------------------------------
// Observation encoder: relu(relu(X @ W1^T + b1) @ W2^T + b2), 32 rows/block
// ---------------------------------------------------------------------------
__global__ void __launch_bounds__(256) obs_kernel(
    const float* __restrict__ vis,
    const __bf16* __restrict__ w1, const float* __restrict__ b1,
    const __bf16* __restrict__ w2, const float* __restrict__ b2,
    __bf16* __restrict__ obsb)
{
  __shared__ __attribute__((aligned(16))) __bf16 sX[32 * VIS];
  __shared__ __attribute__((aligned(16))) float  sF[32 * FF];
  __shared__ __attribute__((aligned(16))) __bf16 sB[32 * FF];
  const int tid = threadIdx.x, wave = tid >> 5, lane = tid & 31;
  const size_t r0 = (size_t)blockIdx.x * 32;

  // vectorized fp32 -> bf16 staging: 2x b128 loads + 1x 16B LDS store / iter
  for (int i = tid * 8; i < 32 * VIS; i += 256 * 8) {
    f32x4 a = *(const f32x4*)(vis + r0 * VIS + i);
    f32x4 b = *(const f32x4*)(vis + r0 * VIS + i + 4);
    bf16x8 o;
    o[0] = (__bf16)a[0]; o[1] = (__bf16)a[1]; o[2] = (__bf16)a[2]; o[3] = (__bf16)a[3];
    o[4] = (__bf16)b[0]; o[5] = (__bf16)b[1]; o[6] = (__bf16)b[2]; o[7] = (__bf16)b[3];
    *(bf16x8*)(sX + i) = o;
  }
  __syncthreads();
  for (int mt = 0; mt < 2; ++mt)
    gemm_tile16(sX + mt * 16 * VIS, VIS, w1, b1, sF + mt * 16 * FF, FF, wave, lane);
  __syncthreads();
  for (int i = tid; i < 32 * FF; i += 256) { float x = sF[i]; sB[i] = (__bf16)(x > 0.f ? x : 0.f); }
  __syncthreads();
  for (int mt = 0; mt < 2; ++mt)
    gemm_tile16(sB + mt * 16 * FF, FF, w2, b2, sF + mt * 16 * FF, FF, wave, lane);
  __syncthreads();
  // vectorized relu + bf16 store (one b128 store per 8 outputs)
  for (int i = tid * 8; i < 32 * FF; i += 256 * 8) {
    bf16x8 o;
    #pragma unroll
    for (int j = 0; j < 8; ++j) { float x = sF[i + j]; o[j] = (__bf16)(x > 0.f ? x : 0.f); }
    *(bf16x8*)(obsb + r0 * FF + i) = o;
  }
}

// ---------------------------------------------------------------------------
// Posterior/prior 2-layer MLP + mu/var heads (block-cooperative, 16 rows)
// ---------------------------------------------------------------------------
static __device__ void mlp_heads(
    const __bf16* sIn, int Kin,
    const __bf16* w1, const float* b1, const __bf16* w2, const float* b2,
    const __bf16* wm, const float* bm, const __bf16* wv, const float* bv,
    float* fT, __bf16* sMid, float* fMu, float* fVar,
    int tid, int wave, int lane)
{
  gemm_tile16(sIn, Kin, w1, b1, fT, FF, wave, lane);
  __syncthreads();
  for (int i = tid; i < 16 * FF; i += 256) { float x = fT[i]; sMid[i] = (__bf16)(x > 0.f ? x : 0.f); }
  __syncthreads();
  gemm_tile16(sMid, FF, w2, b2, fT, FF, wave, lane);
  __syncthreads();
  for (int i = tid; i < 16 * FF; i += 256) { float x = fT[i]; sMid[i] = (__bf16)(x > 0.f ? x : 0.f); }
  __syncthreads();
  gemm_tile16(sMid, FF, wm, bm, fMu, LAT, wave, lane);
  gemm_tile16(sMid, FF, wv, bv, fVar, LAT, wave, lane);
  __syncthreads();
  for (int i = tid; i < 16 * LAT; i += 256) fVar[i] = __expf(fVar[i]) + 0.01f;
  __syncthreads();
}

// ---------------------------------------------------------------------------
// Persistent RSSM scan: 1 block = 16 batch rows through all 127 steps
// ---------------------------------------------------------------------------
__global__ void __launch_bounds__(256) scan_kernel(
    const __bf16* __restrict__ obsb,
    const float* __restrict__ actions,
    const float* __restrict__ noise,
    const __bf16* wgin, const float* bgin,
    const __bf16* wih,  const float* bih,
    const __bf16* whh,  const float* bhh,
    const __bf16* wpr1, const float* bpr1,
    const __bf16* wpr2, const float* bpr2,
    const __bf16* wprm, const float* bprm,
    const __bf16* wprv, const float* bprv,
    const __bf16* wpo1, const float* bpo1,
    const __bf16* wpo2, const float* bpo2,
    const __bf16* wpom, const float* bpom,
    const __bf16* wpov, const float* bpov,
    float* __restrict__ states, float* __restrict__ klout)
{
  __shared__ __attribute__((aligned(16))) __bf16 sSA [16 * KGIN];     // [s | a | pad]
  __shared__ __attribute__((aligned(16))) __bf16 sH  [16 * FF];       // h (bf16 copy)
  __shared__ __attribute__((aligned(16))) __bf16 sG  [16 * FF];
  __shared__ __attribute__((aligned(16))) __bf16 sHO [16 * 2 * FF];   // [h_new | o_t]
  __shared__ __attribute__((aligned(16))) __bf16 sMid[16 * FF];
  __shared__ __attribute__((aligned(16))) float  fH  [16 * FF];       // fp32 h state
  __shared__ __attribute__((aligned(16))) float  fGI [16 * GRU3];     // also fT scratch
  __shared__ __attribute__((aligned(16))) float  fGH [16 * GRU3];     // also mu/var
  __shared__ float fKL[16];

  float* fT   = fGI;                    // gi dead when fT is needed, and vice versa
  float* fMuQ = fGH;                    // gh dead after gates
  float* fVaQ = fGH + 16 * LAT;
  float* fMuP = fGH + 2 * 16 * LAT;
  float* fVaP = fGH + 3 * 16 * LAT;

  const int tid  = threadIdx.x;
  const int wave = tid >> 5, lane = tid & 31;
  const int b0   = blockIdx.x * 16;

  // ---- init: belief=0, h=0, padded [s|a] buffer zeroed, load o_0 ----
  for (int i = tid; i < 16 * KGIN; i += 256) sSA[i] = (__bf16)0.f;
  for (int i = tid; i < 16 * FF; i += 256) { fH[i] = 0.f; sH[i] = (__bf16)0.f; }
  for (int i = tid; i < 16 * FF; i += 256) {
    int r = i >> 8, j = i & 255;
    sHO[r * 512 + j] = (__bf16)0.f;
    sHO[r * 512 + 256 + j] = obsb[((size_t)(b0 + r) * TT + 0) * FF + j];
  }
  __syncthreads();

  // ---- t = 0 posterior -> s0 ----
  mlp_heads(sHO, 2 * FF, wpo1, bpo1, wpo2, bpo2, wpom, bpom, wpov, bpov,
            fT, sMid, fMuQ, fVaQ, tid, wave, lane);
  for (int i = tid; i < 16 * LAT; i += 256) {
    int r = i >> 7, l = i & 127;
    float eps = noise[((size_t)0 * BB + (b0 + r)) * LAT + l];
    float s = fMuQ[i] + sqrtf(fVaQ[i]) * eps;
    states[((size_t)(b0 + r) * TT + 0) * LAT + l] = s;
    sSA[r * KGIN + l] = (__bf16)s;
  }
  __syncthreads();

  for (int t = 1; t < TT; ++t) {
    // Kick off async o_t staging into sHO[:,256:512]; it is not consumed
    // until the posterior MLP, so it overlaps gin/GRU/prior WMMA work.
    // 16 rows x 256 bf16 = 512 x 16B chunks; 2 chunks per thread.
    #pragma unroll
    for (int c2 = 0; c2 < 2; ++c2) {
      int idx = c2 * 256 + tid;          // 0..511
      int r = idx >> 5, c = idx & 31;    // 32 chunks per row
      async_copy_b128(sHO + r * 512 + 256 + c * 8,
                      obsb + ((size_t)(b0 + r) * TT + t) * FF + c * 8);
    }
    // stage a_{t-1}
    for (int i = tid; i < 16 * ACTD; i += 256) {
      int r = i >> 3, c = i & 7;
      sSA[r * KGIN + LAT + c] =
          (__bf16)actions[((size_t)(b0 + r) * (TT - 1) + (t - 1)) * ACTD + c];
    }
    __syncthreads();

    // g = relu([s,a] @ Wgin^T + b)
    gemm_tile16(sSA, KGIN, wgin, bgin, fT, FF, wave, lane);
    __syncthreads();
    for (int i = tid; i < 16 * FF; i += 256) { float x = fT[i]; sG[i] = (__bf16)(x > 0.f ? x : 0.f); }
    __syncthreads();

    // gi = g @ Wih^T + b_ih ; gh = h @ Whh^T + b_hh
    gemm_tile16(sG, FF, wih, bih, fGI, GRU3, wave, lane);
    gemm_tile16(sH, FF, whh, bhh, fGH, GRU3, wave, lane);
    __syncthreads();

    // GRU gates (fp32)
    for (int i = tid; i < 16 * FF; i += 256) {
      int r = i >> 8, j = i & 255;
      float xr = fGI[r * GRU3 + j]          + fGH[r * GRU3 + j];
      float xz = fGI[r * GRU3 + FF + j]     + fGH[r * GRU3 + FF + j];
      float xn = fGI[r * GRU3 + 2 * FF + j];
      float hn = fGH[r * GRU3 + 2 * FF + j];
      float rg = 1.f / (1.f + __expf(-xr));
      float zg = 1.f / (1.f + __expf(-xz));
      float ng = tanhf(xn + rg * hn);
      float h  = (1.f - zg) * ng + zg * fH[i];
      fH[i] = h;
      __bf16 hb = (__bf16)h;
      sH[i] = hb;
      sHO[r * 512 + j] = hb;
    }
    __syncthreads();

    // prior heads (fGH reused for mu/var after gates)
    mlp_heads(sH, FF, wpr1, bpr1, wpr2, bpr2, wprm, bprm, wprv, bprv,
              fT, sMid, fMuP, fVaP, tid, wave, lane);

    // drain async o_t copy, publish to the block, then posterior heads
    async_wait0();
    __syncthreads();
    mlp_heads(sHO, 2 * FF, wpo1, bpo1, wpo2, bpo2, wpom, bpom, wpov, bpov,
              fT, sMid, fMuQ, fVaQ, tid, wave, lane);

    // reparameterized sample + state output
    for (int i = tid; i < 16 * LAT; i += 256) {
      int r = i >> 7, l = i & 127;
      float eps = noise[((size_t)t * BB + (b0 + r)) * LAT + l];
      float s = fMuQ[i] + sqrtf(fVaQ[i]) * eps;
      states[((size_t)(b0 + r) * TT + t) * LAT + l] = s;
      sSA[r * KGIN + l] = (__bf16)s;
    }
    // KL reduction over LAT per row
    if (tid < 16) fKL[tid] = 0.f;
    __syncthreads();
    for (int i = tid; i < 16 * LAT; i += 256) {
      int r = i >> 7;
      float vp = fVaP[i], vq = fVaQ[i];
      float dm = fMuQ[i] - fMuP[i];
      float term = __logf(vp) - __logf(vq) + (vq + dm * dm) / vp - 1.f;
      atomicAdd(&fKL[r], term);
    }
    __syncthreads();
    if (tid < 16) klout[(size_t)(b0 + tid) * (TT - 1) + (t - 1)] = 0.5f * fKL[tid];
    __syncthreads();
  }
}

// ---------------------------------------------------------------------------
// Workspace layout (bf16 elements), all 16B-aligned offsets
// ---------------------------------------------------------------------------
static constexpr size_t O_OBS1 = 0;                       // 256*512
static constexpr size_t O_OBS2 = O_OBS1 + 131072;         // 256*256
static constexpr size_t O_GIN  = O_OBS2 + 65536;          // 256*160 (padded)
static constexpr size_t O_IH   = O_GIN  + 40960;          // 768*256
static constexpr size_t O_HH   = O_IH   + 196608;         // 768*256
static constexpr size_t O_PR1  = O_HH   + 196608;         // 256*256
static constexpr size_t O_PR2  = O_PR1  + 65536;          // 256*256
static constexpr size_t O_PRM  = O_PR2  + 65536;          // 128*256
static constexpr size_t O_PRV  = O_PRM  + 32768;          // 128*256
static constexpr size_t O_PO1  = O_PRV  + 32768;          // 256*512
static constexpr size_t O_PO2  = O_PO1  + 131072;         // 256*256
static constexpr size_t O_POM  = O_PO2  + 65536;          // 128*256
static constexpr size_t O_POV  = O_POM  + 32768;          // 128*256
static constexpr size_t O_OBSB = O_POV  + 32768;          // 512*128*256 bf16 obs

extern "C" void kernel_launch(void* const* d_in, const int* in_sizes, int n_in,
                              void* d_out, int out_size, void* d_ws, size_t ws_size,
                              hipStream_t stream)
{
  (void)in_sizes; (void)n_in; (void)out_size; (void)ws_size;
  const float* vis    = (const float*)d_in[0];
  const float* acts   = (const float*)d_in[1];
  const float* noise  = (const float*)d_in[2];
  const float* w_obs1 = (const float*)d_in[3];
  const float* b_obs1 = (const float*)d_in[4];
  const float* w_obs2 = (const float*)d_in[5];
  const float* b_obs2 = (const float*)d_in[6];
  const float* w_gin  = (const float*)d_in[7];
  const float* b_gin  = (const float*)d_in[8];
  const float* w_ih   = (const float*)d_in[9];
  const float* w_hh   = (const float*)d_in[10];
  const float* b_ih   = (const float*)d_in[11];
  const float* b_hh   = (const float*)d_in[12];
  const float* w_pr1  = (const float*)d_in[13];
  const float* b_pr1  = (const float*)d_in[14];
  const float* w_pr2  = (const float*)d_in[15];
  const float* b_pr2  = (const float*)d_in[16];
  const float* w_prm  = (const float*)d_in[17];
  const float* b_prm  = (const float*)d_in[18];
  const float* w_prv  = (const float*)d_in[19];
  const float* b_prv  = (const float*)d_in[20];
  const float* w_po1  = (const float*)d_in[21];
  const float* b_po1  = (const float*)d_in[22];
  const float* w_po2  = (const float*)d_in[23];
  const float* b_po2  = (const float*)d_in[24];
  const float* w_pom  = (const float*)d_in[25];
  const float* b_pom  = (const float*)d_in[26];
  const float* w_pov  = (const float*)d_in[27];
  const float* b_pov  = (const float*)d_in[28];

  __bf16* ws = (__bf16*)d_ws;
  auto cvt = [&](const float* src, size_t off, int n) {
    cvt_bf16<<<(n + 255) / 256, 256, 0, stream>>>(src, ws + off, n);
  };
  cvt(w_obs1, O_OBS1, FF * VIS);
  cvt(w_obs2, O_OBS2, FF * FF);
  cvt(w_ih,   O_IH,   GRU3 * FF);
  cvt(w_hh,   O_HH,   GRU3 * FF);
  cvt(w_pr1,  O_PR1,  FF * FF);
  cvt(w_pr2,  O_PR2,  FF * FF);
  cvt(w_prm,  O_PRM,  LAT * FF);
  cvt(w_prv,  O_PRV,  LAT * FF);
  cvt(w_po1,  O_PO1,  FF * 2 * FF);
  cvt(w_po2,  O_PO2,  FF * FF);
  cvt(w_pom,  O_POM,  LAT * FF);
  cvt(w_pov,  O_POV,  LAT * FF);
  cvt_gin<<<(FF * KGIN + 255) / 256, 256, 0, stream>>>(w_gin, ws + O_GIN);

  obs_kernel<<<(BB * TT) / 32, 256, 0, stream>>>(
      vis, ws + O_OBS1, b_obs1, ws + O_OBS2, b_obs2, ws + O_OBSB);

  float* states = (float*)d_out;
  float* klout  = states + (size_t)BB * TT * LAT;
  scan_kernel<<<BB / 16, 256, 0, stream>>>(
      ws + O_OBSB, acts, noise,
      ws + O_GIN, b_gin, ws + O_IH, b_ih, ws + O_HH, b_hh,
      ws + O_PR1, b_pr1, ws + O_PR2, b_pr2, ws + O_PRM, b_prm, ws + O_PRV, b_prv,
      ws + O_PO1, b_po1, ws + O_PO2, b_po2, ws + O_POM, b_pom, ws + O_POV, b_pov,
      states, klout);
}
